// Transformer_GNN_32341103739534
// MI455X (gfx1250) — matrix operand order repaired and checked by reading, not compiled, not run on hardware
//
#include <hip/hip_runtime.h>
#include <stdint.h>
#include <stddef.h>

// Problem constants (match reference)
#define NN    50000
#define EE    800000
#define HEADS 8

typedef __attribute__((ext_vector_type(16))) __bf16 v16bf;
typedef __attribute__((ext_vector_type(8)))  __bf16 v8bf;
typedef __attribute__((ext_vector_type(8)))  float  v8f;

__device__ __forceinline__ unsigned short f32_to_bf16(float f) {
  unsigned int u = __float_as_uint(f);
  u += 0x7FFFu + ((u >> 16) & 1u);           // round-to-nearest-even
  return (unsigned short)(u >> 16);
}

// ---------------- prep kernels ----------------

__global__ void __launch_bounds__(256)
k_to_bf16(const float* __restrict__ in, unsigned short* __restrict__ out, int n) {
  int t = blockIdx.x * blockDim.x + threadIdx.x;
  if (t < n) out[t] = f32_to_bf16(in[t]);
}

// gelu(tanh approx, jax.nn.gelu default) + convert to bf16
__global__ void __launch_bounds__(256)
k_gelu_bf16(const float* __restrict__ in, unsigned short* __restrict__ out, int n) {
  int t = blockIdx.x * blockDim.x + threadIdx.x;
  if (t >= n) return;
  float x = in[t];
  float inner = 0.7978845608028654f * (x + 0.044715f * x * x * x);
  float g = 0.5f * x * (1.0f + tanhf(inner));
  out[t] = f32_to_bf16(g);
}

// W[K x 128] f32 -> Wt[128 x K] bf16 (so B-fragments are contiguous in K)
__global__ void __launch_bounds__(256)
k_wt(const float* __restrict__ W, unsigned short* __restrict__ Wt, int K) {
  int t = blockIdx.x * blockDim.x + threadIdx.x;
  if (t >= K * 128) return;
  int k = t >> 7, c = t & 127;
  Wt[(size_t)c * K + k] = f32_to_bf16(W[(size_t)k * 128 + c]);
}

__global__ void __launch_bounds__(256)
k_init_seg(unsigned int* __restrict__ mkey, float* __restrict__ denom, int n) {
  int t = blockIdx.x * blockDim.x + threadIdx.x;
  if (t < n) { mkey[t] = 0u; denom[t] = 0.0f; }
}

// ---- fused 4-matrix WMMA GEMM: Ym[N x 128] = Xb[N x K](bf16) @ Wm + bm ----
// Wt* are W^T as bf16 [128 x K]. One wave per 16-row strip; A fragments loaded
// once, reused across 4 weight matrices x 8 column tiles (up to 128 WMMAs per
// wave). Column tiles are processed in PAIRS with two independent accumulators
// so the scheduler always has an independent WMMA/store/splat stream to fill
// the WMMA->VALU hazard slots, instead of v_nops.
template <int K>
__global__ void __launch_bounds__(256)
k_gemm4_wmma(const unsigned short* __restrict__ Xb,
             const unsigned short* __restrict__ Wt0,
             const unsigned short* __restrict__ Wt1,
             const unsigned short* __restrict__ Wt2,
             const unsigned short* __restrict__ Wt3,
             const float* __restrict__ b0, const float* __restrict__ b1,
             const float* __restrict__ b2, const float* __restrict__ b3,
             float* __restrict__ Y0, float* __restrict__ Y1,
             float* __restrict__ Y2, float* __restrict__ Y3,
             int nrows) {
  constexpr int KS = K / 32;
  const int lane = threadIdx.x & 31;
  const int rowTile = blockIdx.x * 8 + (threadIdx.x >> 5);
  const int row0 = rowTile << 4;
  if (row0 >= nrows) return;                 // wave-uniform exit (EXEC stays all-1)
  const int hi = lane >> 4;                  // lane half selects K sub-block
  const int lm = lane & 15;                  // row (A) / col (B,C)

  // A fragments: lane L holds row row0+lm, K = 16*hi..16*hi+15 (contiguous)
  v16bf afrag[KS];
  const unsigned short* arow = Xb + (size_t)(row0 + lm) * K;
#pragma unroll
  for (int s = 0; s < KS; ++s) {
    afrag[s] = *(const v16bf*)(arow + s * 32 + hi * 16);
  }

  const unsigned short* wts[4]  = {Wt0, Wt1, Wt2, Wt3};
  const float*          bia[4]  = {b0, b1, b2, b3};
  float*                yout[4] = {Y0, Y1, Y2, Y3};

#pragma unroll
  for (int m = 0; m < 4; ++m) {
    const unsigned short* Wt = wts[m];
    const float* bias = bia[m];
    float* Y = yout[m];

#pragma unroll
    for (int cp = 0; cp < 4; ++cp) {         // pair of column tiles: cols cp*32 +0/+16
      const int c0 = cp * 32 + lm;
      const int c1 = c0 + 16;

      // B fragment: lane holds col c, K = s*32 + 16*hi .. +15 (contiguous 32B)
      v16bf bf0[KS], bf1[KS];
      const unsigned short* bcol0 = Wt + (size_t)c0 * K;
      const unsigned short* bcol1 = Wt + (size_t)c1 * K;
#pragma unroll
      for (int s = 0; s < KS; ++s) {
        bf0[s] = *(const v16bf*)(bcol0 + s * 32 + hi * 16);
        bf1[s] = *(const v16bf*)(bcol1 + s * 32 + hi * 16);
      }

      const float bv0 = bias[c0];
      const float bv1 = bias[c1];
      v8f acc0, acc1;
#pragma unroll
      for (int r = 0; r < 8; ++r) { acc0[r] = bv0; acc1[r] = bv1; }

#pragma unroll
      for (int s = 0; s < KS; ++s) {         // interleave two independent chains
        acc0 = __builtin_amdgcn_wmma_f32_16x16x32_bf16(
            false, afrag[s], false, bf0[s], (short)0, acc0, false, false);
        acc1 = __builtin_amdgcn_wmma_f32_16x16x32_bf16(
            false, afrag[s], false, bf1[s], (short)0, acc1, false, false);
      }

      // C/D layout: VGPR r -> row r+8*hi, col lm
      float* yb0 = Y + (size_t)row0 * 128 + c0;
      float* yb1 = Y + (size_t)row0 * 128 + c1;
#pragma unroll
      for (int r = 0; r < 8; ++r) {
        yb0[(size_t)(r + 8 * hi) * 128] = acc0[r];
        yb1[(size_t)(r + 8 * hi) * 128] = acc1[r];
      }
    }
  }
}

// ---------------- edge phase ----------------
// monotonic float->uint key so segment-max uses native u32 atomic max
__device__ __forceinline__ unsigned int f2key(float f) {
  unsigned int u = __float_as_uint(f);
  return (u & 0x80000000u) ? ~u : (u | 0x80000000u);
}
__device__ __forceinline__ float key2f(unsigned int k) {
  unsigned int u = (k & 0x80000000u) ? (k ^ 0x80000000u) : ~k;
  return __uint_as_float(u);
}

// pass 1: alpha[e,h] = <q[dst,h,:], k[src,h,:]> / 4 ; segment max over dst
__global__ void __launch_bounds__(256)
k_edge_logits(const long long* __restrict__ ei, const float* __restrict__ Q,
              const float* __restrict__ Kf, float* __restrict__ alpha,
              unsigned int* __restrict__ mkey) {
  int t = blockIdx.x * blockDim.x + threadIdx.x;
  if (t >= EE * HEADS) return;
  int e = t >> 3, h = t & 7;
  int src = (int)ei[e];
  int dst = (int)ei[EE + e];
  const float4* q = (const float4*)(Q + (size_t)dst * 128 + h * 16);
  const float4* k = (const float4*)(Kf + (size_t)src * 128 + h * 16);
  float s = 0.0f;
#pragma unroll
  for (int i = 0; i < 4; ++i) {
    float4 a = q[i], b = k[i];
    s += a.x * b.x + a.y * b.y + a.z * b.z + a.w * b.w;
  }
  s *= 0.25f;  // 1/sqrt(16)
  alpha[t] = s;
  atomicMax(mkey + (size_t)dst * HEADS + h, f2key(s));
}

// pass 2: ex = exp(alpha - max); segment sum over dst
__global__ void __launch_bounds__(256)
k_edge_exp(const long long* __restrict__ ei, float* __restrict__ alpha,
           const unsigned int* __restrict__ mkey, float* __restrict__ denom) {
  int t = blockIdx.x * blockDim.x + threadIdx.x;
  if (t >= EE * HEADS) return;
  int e = t >> 3, h = t & 7;
  int dst = (int)ei[EE + e];
  float m = key2f(mkey[(size_t)dst * HEADS + h]);
  float ex = __expf(alpha[t] - m);
  alpha[t] = ex;
  atomicAdd(denom + (size_t)dst * HEADS + h, ex);
}

// pass 3: out[dst] += (ex/denom) * v[src]  (out pre-filled with x@Ws + bs)
// one thread per (edge, head): w computed once, 4x float4 gather + 16 atomics
__global__ void __launch_bounds__(256)
k_edge_scatter(const long long* __restrict__ ei, const float* __restrict__ ex,
               const float* __restrict__ denom, const float* __restrict__ V,
               float* __restrict__ out) {
  int t = blockIdx.x * blockDim.x + threadIdx.x;
  if (t >= EE * HEADS) return;
  int e = t >> 3, h = t & 7;
  int src = (int)ei[e];
  int dst = (int)ei[EE + e];
  float w = ex[(size_t)e * HEADS + h] / denom[(size_t)dst * HEADS + h];
  const float4* v = (const float4*)(V + (size_t)src * 128 + h * 16);
  float* o = out + (size_t)dst * 128 + h * 16;
#pragma unroll
  for (int i = 0; i < 4; ++i) {
    float4 vv = v[i];
    atomicAdd(o + i * 4 + 0, w * vv.x);
    atomicAdd(o + i * 4 + 1, w * vv.y);
    atomicAdd(o + i * 4 + 2, w * vv.z);
    atomicAdd(o + i * 4 + 3, w * vv.w);
  }
}

// ---------------- host orchestration ----------------

static inline int gr(int n) { return (n + 255) / 256; }

extern "C" void kernel_launch(void* const* d_in, const int* in_sizes, int n_in,
                              void* d_out, int out_size, void* d_ws, size_t ws_size,
                              hipStream_t stream) {
  const float*     x   = (const float*)d_in[0];
  const long long* ei  = (const long long*)d_in[1];  // int64 [2, E]
  const float* Wq1 = (const float*)d_in[2];
  const float* Wk1 = (const float*)d_in[3];
  const float* Wv1 = (const float*)d_in[4];
  const float* Ws1 = (const float*)d_in[5];
  const float* bq1 = (const float*)d_in[6];
  const float* bk1 = (const float*)d_in[7];
  const float* bv1 = (const float*)d_in[8];
  const float* bs1 = (const float*)d_in[9];
  const float* Wq2 = (const float*)d_in[10];
  const float* Wk2 = (const float*)d_in[11];
  const float* Wv2 = (const float*)d_in[12];
  const float* Ws2 = (const float*)d_in[13];
  const float* bq2 = (const float*)d_in[14];
  const float* bk2 = (const float*)d_in[15];
  const float* bv2 = (const float*)d_in[16];
  const float* bs2 = (const float*)d_in[17];
  float* out = (float*)d_out;

  // workspace carve (all re-initialized/overwritten every call; ~145 MB,
  // small enough that the hot set lives in the 192 MB L2)
  char* p = (char*)d_ws;
  auto carve = [&](size_t bytes) -> void* {
    void* r = (void*)p; p += (bytes + 255) & ~(size_t)255; return r;
  };
  unsigned short* xb    = (unsigned short*)carve((size_t)NN * 128 * 2);
  unsigned short* wtq   = (unsigned short*)carve((size_t)128 * 128 * 2);
  unsigned short* wtk   = (unsigned short*)carve((size_t)128 * 128 * 2);
  unsigned short* wtv   = (unsigned short*)carve((size_t)128 * 128 * 2);
  unsigned short* wts   = (unsigned short*)carve((size_t)128 * 128 * 2);
  float*        Q     = (float*)carve((size_t)NN * 128 * 4);
  float*        Kf    = (float*)carve((size_t)NN * 128 * 4);
  float*        V     = (float*)carve((size_t)NN * 128 * 4);
  float*        h1    = (float*)carve((size_t)NN * 128 * 4);
  float*        alpha = (float*)carve((size_t)EE * HEADS * 4);
  unsigned int* mkey  = (unsigned int*)carve((size_t)NN * HEADS * 4);
  float*        denom = (float*)carve((size_t)NN * HEADS * 4);

  const int rowBlocks = ((NN / 16) + 7) / 8;  // 8 waves (row tiles) per block

  // ---------------- layer 1 (K = 64) ----------------
  k_to_bf16<<<gr(NN * 64), 256, 0, stream>>>(x, xb, NN * 64);
  k_wt<<<gr(64 * 128), 256, 0, stream>>>(Wq1, wtq, 64);
  k_wt<<<gr(64 * 128), 256, 0, stream>>>(Wk1, wtk, 64);
  k_wt<<<gr(64 * 128), 256, 0, stream>>>(Wv1, wtv, 64);
  k_wt<<<gr(64 * 128), 256, 0, stream>>>(Ws1, wts, 64);

  k_gemm4_wmma<64><<<rowBlocks, 256, 0, stream>>>(
      xb, wtq, wtk, wtv, wts, bq1, bk1, bv1, bs1, Q, Kf, V, h1, NN);

  k_init_seg<<<gr(NN * HEADS), 256, 0, stream>>>(mkey, denom, NN * HEADS);
  k_edge_logits<<<gr(EE * HEADS), 256, 0, stream>>>(ei, Q, Kf, alpha, mkey);
  k_edge_exp<<<gr(EE * HEADS), 256, 0, stream>>>(ei, alpha, mkey, denom);
  k_edge_scatter<<<gr(EE * HEADS), 256, 0, stream>>>(ei, alpha, denom, V, h1);

  // ---------------- layer 2 (K = 128) ----------------
  k_gelu_bf16<<<gr(NN * 128), 256, 0, stream>>>(h1, xb, NN * 128);
  k_wt<<<gr(128 * 128), 256, 0, stream>>>(Wq2, wtq, 128);
  k_wt<<<gr(128 * 128), 256, 0, stream>>>(Wk2, wtk, 128);
  k_wt<<<gr(128 * 128), 256, 0, stream>>>(Wv2, wtv, 128);
  k_wt<<<gr(128 * 128), 256, 0, stream>>>(Ws2, wts, 128);

  k_gemm4_wmma<128><<<rowBlocks, 256, 0, stream>>>(
      xb, wtq, wtk, wtv, wts, bq2, bk2, bv2, bs2, Q, Kf, V, out, NN);

  k_init_seg<<<gr(NN * HEADS), 256, 0, stream>>>(mkey, denom, NN * HEADS);
  k_edge_logits<<<gr(EE * HEADS), 256, 0, stream>>>(ei, Q, Kf, alpha, mkey);
  k_edge_exp<<<gr(EE * HEADS), 256, 0, stream>>>(ei, alpha, mkey, denom);
  k_edge_scatter<<<gr(EE * HEADS), 256, 0, stream>>>(ei, alpha, denom, V, out);
}
